// CBFGNN_20315195310656
// MI455X (gfx1250) — compile-verified, matrix operand order checked by reading, not run on hardware
//
#include <hip/hip_runtime.h>
#include <hip/hip_bf16.h>
#include <stdint.h>

// ---------------------------------------------------------------------------
// CBF-GNN forward on MI455X (gfx1250): per-edge / per-node MLPs as wave32
// 16-row WMMA GEMM tiles (v_wmma_f32_16x16x32_f16), scatter-max via
// global_atomic_max_num_f32.
// ---------------------------------------------------------------------------

typedef _Float16 v16h __attribute__((ext_vector_type(16)));
typedef _Float16 v8h  __attribute__((ext_vector_type(8)));
typedef _Float16 v4h  __attribute__((ext_vector_type(4)));
typedef float    v8f  __attribute__((ext_vector_type(8)));

#define N_NODES 65536
#define N_EDGES 1048576

// -------------------- workspace layout (bytes) -----------------------------
static const size_t OFF_AGG1 = 0;                                   // 32 MB
static const size_t OFF_AGG2 = OFF_AGG1 + (size_t)N_NODES * 128 * 4; // 32 MB
static const size_t OFF_H1   = OFF_AGG2 + (size_t)N_NODES * 128 * 4; // 16 MB
static const size_t OFF_WF   = OFF_H1   + (size_t)N_NODES * 64 * 4;  // frags

// weight-fragment offsets in halves (each fragment = 32 lanes * 16 halves)
enum {
  WF_P1_0 = 0,       // 96x64   KB=3 NT=4  -> 6144
  WF_P1_1 = 6144,    // 64x64   KB=2 NT=4  -> 4096
  WF_P1_2 = 10240,   // 64x128  KB=2 NT=8  -> 8192
  WF_G1_0 = 18432,   // 160x64  KB=5 NT=4  -> 10240
  WF_G1_1 = 28672,   // 64x64
  WF_G1_2 = 32768,   // 64x64
  WF_P2_0 = 36864,   // 160x64
  WF_P2_1 = 47104,   // 64x64
  WF_P2_2 = 51200,   // 64x128
  WF_G2_0 = 59392,   // 192x64  KB=6 NT=4  -> 12288
  WF_G2_1 = 71680,   // 64x64
  WF_G2_2 = 75776,   // 64x64
  WF_H_0  = 79872,   // 64x64
  WF_H_1  = 83968,   // 64x64
  WF_TOTAL = 88064
};

// -------------------- WMMA helpers -----------------------------------------
__device__ __forceinline__ v8f wmma32(v16h a, v16h b, v8f c) {
  return __builtin_amdgcn_wmma_f32_16x16x32_f16(
      /*neg_a=*/false, a, /*neg_b=*/false, b,
      /*c_mod=*/(short)0, c, /*reuse_a=*/false, /*reuse_b=*/false);
}

// A fragment for one 32-wide K block of a 16xK f16 tile stored row-major in LDS.
// ISA layout: lanes 0-15 hold row (lane&15) halves K={0..7,16..23}(+32*kb),
// lanes 16-31 hold K={8..15,24..31}(+32*kb).
__device__ __forceinline__ v16h load_afrag(const _Float16* tile, int K, int kb, int lane) {
  const int m = lane & 15;
  const int base_k = kb * 32 + ((lane & 16) ? 8 : 0);
  const _Float16* row = tile + m * K;
  v8h lo = *(const v8h*)(row + base_k);
  v8h hi = *(const v8h*)(row + base_k + 16);
  v16h a;
#pragma unroll
  for (int i = 0; i < 8; ++i) { a[i] = lo[i]; a[i + 8] = hi[i]; }
  return a;
}

// B fragment precomputed in lane-major order: 32B contiguous per lane.
__device__ __forceinline__ v16h load_bfrag(const _Float16* __restrict__ wf,
                                           int kb, int nt, int NT, int lane) {
  return *(const v16h*)(wf + ((size_t)(kb * NT + nt) * 32 + lane) * 16);
}

// C/D epilogue: bias (+ReLU) and store as f16 into a 16x64 row-major LDS tile.
__device__ __forceinline__ void store_h16(v8f acc, _Float16* dsth,
                                          const float* __restrict__ bias,
                                          int nt, int lane, bool relu) {
  const int n = nt * 16 + (lane & 15);
  const float b = bias[n];
  const int mo = (lane & 16) ? 8 : 0;
#pragma unroll
  for (int r = 0; r < 8; ++r) {
    float v = acc[r] + b;
    if (relu) v = fmaxf(v, 0.f);
    dsth[(r + mo) * 64 + n] = (_Float16)v;
  }
}

__device__ __forceinline__ void cvt_store4(_Float16* dst, float4 f) {
  v4h h; h[0] = (_Float16)f.x; h[1] = (_Float16)f.y;
  h[2] = (_Float16)f.z; h[3] = (_Float16)f.w;
  *(v4h*)dst = h;
}
// clamp -inf (empty segment) to 0, per reference isfinite handling
__device__ __forceinline__ void cvt_store4c(_Float16* dst, float4 f) {
  v4h h;
  h[0] = (_Float16)(f.x < -1e37f ? 0.f : f.x);
  h[1] = (_Float16)(f.y < -1e37f ? 0.f : f.y);
  h[2] = (_Float16)(f.z < -1e37f ? 0.f : f.z);
  h[3] = (_Float16)(f.w < -1e37f ? 0.f : f.w);
  *(v4h*)dst = h;
}

__device__ __forceinline__ void atom_max_f32(float* p, float v) {
  asm volatile("global_atomic_max_num_f32 %0, %1, off"
               :: "v"((unsigned long long)(uintptr_t)p), "v"(v)
               : "memory");
}

// One 16xK @ KxN(64) GEMM stage: A from LDS, B frags from global, f16 out to LDS.
template <int KB, int NT>
__device__ __forceinline__ void gemm_stage(const _Float16* src, int K,
                                           const _Float16* __restrict__ wf,
                                           const float* __restrict__ bias,
                                           bool relu, _Float16* dsth, int lane) {
  __syncthreads();  // previous-stage LDS writes visible
  v16h a[KB];
#pragma unroll
  for (int kb = 0; kb < KB; ++kb) a[kb] = load_afrag(src, K, kb, lane);
#pragma unroll
  for (int nt = 0; nt < NT; ++nt) {
    v8f acc = {};
#pragma unroll
    for (int kb = 0; kb < KB; ++kb)
      acc = wmma32(a[kb], load_bfrag(wf, kb, nt, NT, lane), acc);
    store_h16(acc, dsth, bias, nt, lane, relu);
  }
}

// -------------------- utility kernels --------------------------------------
__global__ void k_fill(float* p, size_t n, float v) {
  size_t i = (size_t)blockIdx.x * blockDim.x + threadIdx.x;
  const size_t stride = (size_t)gridDim.x * blockDim.x;
  for (; i < n; i += stride) p[i] = v;
}

// Convert row-major f32 W[K][N] into WMMA-B lane-major f16 fragments.
__global__ void k_wfrag(const float* __restrict__ W, _Float16* __restrict__ dst,
                        int N, int NT) {
  const int f = blockIdx.x;            // kb*NT + nt
  const int kb = f / NT, nt = f % NT;
  const int t = threadIdx.x;           // 0..511
  const int lane = t >> 4, i = t & 15;
  const int klocal = (i < 8 ? i : i + 8) + ((lane & 16) ? 8 : 0);
  const int k = kb * 32 + klocal;
  const int n = nt * 16 + (lane & 15);
  dst[(size_t)f * 512 + lane * 16 + i] = (_Float16)W[(size_t)k * N + n];
}

// -------------------- edge message (phi) kernel -----------------------------
// XDIM: node-feature width (32 layer1, 64 layer2); KIN = 2*XDIM+32; KB_IN = KIN/32
template <int XDIM, int KIN, int KB_IN>
__global__ void __launch_bounds__(128)
k_phi(const float* __restrict__ xin, const float* __restrict__ eattr,
      const long long* __restrict__ ei,
      const _Float16* __restrict__ wf0, const _Float16* __restrict__ wf1,
      const _Float16* __restrict__ wf2,
      const float* __restrict__ b0, const float* __restrict__ b1,
      const float* __restrict__ b2,
      float* __restrict__ agg) {
  const int wid = threadIdx.x >> 5, lane = threadIdx.x & 31;
  const int tile = blockIdx.x * 4 + wid;
  __shared__ __attribute__((aligned(32))) _Float16 sh[4][16 * KIN + 2048];
  __shared__ int sdst[4][16];
  _Float16* base = sh[wid];
  _Float16* hA = base + 16 * KIN;
  _Float16* hB = hA + 1024;

  { // gather: row = [x[dst] | x[src] | edge_attr] as f16
    const int part = lane >> 4, r = lane & 15;
    const long long e = (long long)tile * 16 + r;
    const long long sidx = ei[e];
    const long long didx = ei[(long long)N_EDGES + e];
    if (!part) sdst[wid][r] = (int)didx;
    const float* np = xin + (size_t)(part ? sidx : didx) * XDIM;
    _Float16* row = base + r * KIN;
#pragma unroll
    for (int j = 0; j < XDIM; j += 4)
      cvt_store4(row + part * XDIM + j, *(const float4*)(np + j));
    const float* ap = eattr + e * 32 + part * 16;
#pragma unroll
    for (int j = 0; j < 16; j += 4)
      cvt_store4(row + 2 * XDIM + part * 16 + j, *(const float4*)(ap + j));
  }

  gemm_stage<KB_IN, 4>(base, KIN, wf0, b0, true, hA, lane);  // KIN -> 64, ReLU
  gemm_stage<2, 4>(hA, 64, wf1, b1, true, hB, lane);         // 64 -> 64, ReLU

  // final 64 -> 128 + scatter-max into agg[dst]
  __syncthreads();
  v16h a0 = load_afrag(hB, 64, 0, lane);
  v16h a1 = load_afrag(hB, 64, 1, lane);
  const int mo = (lane & 16) ? 8 : 0;
#pragma unroll
  for (int nt = 0; nt < 8; ++nt) {
    v8f acc = {};
    acc = wmma32(a0, load_bfrag(wf2, 0, nt, 8, lane), acc);
    acc = wmma32(a1, load_bfrag(wf2, 1, nt, 8, lane), acc);
    const int n = nt * 16 + (lane & 15);
    const float bb = b2[n];
#pragma unroll
    for (int r = 0; r < 8; ++r) {
      float v = acc[r] + bb;
      atom_max_f32(agg + (size_t)sdst[wid][r + mo] * 128 + n, v);
    }
  }
}

// -------------------- node update gamma layer 1 ------------------------------
__global__ void __launch_bounds__(128)
k_gamma1(const float* __restrict__ agg, const float* __restrict__ x,
         const _Float16* __restrict__ wf0, const _Float16* __restrict__ wf1,
         const _Float16* __restrict__ wf2,
         const float* __restrict__ b0, const float* __restrict__ b1,
         const float* __restrict__ b2,
         float* __restrict__ h1out) {
  const int wid = threadIdx.x >> 5, lane = threadIdx.x & 31;
  const int tile = blockIdx.x * 4 + wid;
  __shared__ __attribute__((aligned(32))) _Float16 sh[4][16 * 160 + 2048];
  _Float16* base = sh[wid];
  _Float16* hA = base + 16 * 160;
  _Float16* hB = hA + 1024;

  { // gather: [agg (128, clamped) | x (32)]
    const int part = lane >> 4, r = lane & 15;
    const size_t node = (size_t)tile * 16 + r;
    _Float16* row = base + r * 160;
    const float* ag = agg + node * 128 + part * 64;
#pragma unroll
    for (int j = 0; j < 64; j += 4)
      cvt_store4c(row + part * 64 + j, *(const float4*)(ag + j));
    const float* xp = x + node * 32 + part * 16;
#pragma unroll
    for (int j = 0; j < 16; j += 4)
      cvt_store4(row + 128 + part * 16 + j, *(const float4*)(xp + j));
  }

  gemm_stage<5, 4>(base, 160, wf0, b0, true, hA, lane);
  gemm_stage<2, 4>(hA, 64, wf1, b1, true, hB, lane);

  // final 64 -> 64, then relu(h) between layers; store f32 to global
  __syncthreads();
  v16h a0 = load_afrag(hB, 64, 0, lane);
  v16h a1 = load_afrag(hB, 64, 1, lane);
  const int mo = (lane & 16) ? 8 : 0;
#pragma unroll
  for (int nt = 0; nt < 4; ++nt) {
    v8f acc = {};
    acc = wmma32(a0, load_bfrag(wf2, 0, nt, 4, lane), acc);
    acc = wmma32(a1, load_bfrag(wf2, 1, nt, 4, lane), acc);
    const int n = nt * 16 + (lane & 15);
    const float bb = b2[n];
#pragma unroll
    for (int r = 0; r < 8; ++r) {
      const float v = fmaxf(acc[r] + bb, 0.f);
      h1out[(size_t)(tile * 16 + r + mo) * 64 + n] = v;
    }
  }
}

// -------------------- gamma layer 2 + head ----------------------------------
__global__ void __launch_bounds__(128)
k_gamma2_head(const float* __restrict__ agg, const float* __restrict__ h1,
              const _Float16* __restrict__ g0, const _Float16* __restrict__ g1,
              const _Float16* __restrict__ g2,
              const float* __restrict__ gb0, const float* __restrict__ gb1,
              const float* __restrict__ gb2,
              const _Float16* __restrict__ hw0, const _Float16* __restrict__ hw1,
              const float* __restrict__ hb0, const float* __restrict__ hb1,
              const float* __restrict__ w2, const float* __restrict__ b2s,
              float* __restrict__ out) {
  const int wid = threadIdx.x >> 5, lane = threadIdx.x & 31;
  const int tile = blockIdx.x * 4 + wid;
  __shared__ __attribute__((aligned(32))) _Float16 sh[4][16 * 192 + 2048];
  _Float16* base = sh[wid];
  _Float16* hA = base + 16 * 192;
  _Float16* hB = hA + 1024;

  { // gather: [agg2 (128, clamped) | h1 (64)]
    const int part = lane >> 4, r = lane & 15;
    const size_t node = (size_t)tile * 16 + r;
    _Float16* row = base + r * 192;
    const float* ag = agg + node * 128 + part * 64;
#pragma unroll
    for (int j = 0; j < 64; j += 4)
      cvt_store4c(row + part * 64 + j, *(const float4*)(ag + j));
    const float* hp = h1 + node * 64 + part * 32;
#pragma unroll
    for (int j = 0; j < 32; j += 4)
      cvt_store4(row + 128 + part * 32 + j, *(const float4*)(hp + j));
  }

  gemm_stage<6, 4>(base, 192, g0, gb0, true,  hA, lane);  // gamma 192->64 ReLU
  gemm_stage<2, 4>(hA, 64,  g1, gb1, true,  hB, lane);    // gamma 64->64 ReLU
  gemm_stage<2, 4>(hB, 64,  g2, gb2, false, hA, lane);    // gamma 64->64 (no act)
  gemm_stage<2, 4>(hA, 64, hw0, hb0, true,  hB, lane);    // head 64->64 ReLU
  gemm_stage<2, 4>(hB, 64, hw1, hb1, true,  hA, lane);    // head 64->64 ReLU

  // head final 64 -> 1 dot product (split across lane halves, shfl-reduce)
  __syncthreads();
  const int part = lane >> 4, m = lane & 15;
  float s = 0.f;
#pragma unroll
  for (int k = 0; k < 32; ++k)
    s += (float)hA[m * 64 + part * 32 + k] * w2[part * 32 + k];
  s += __shfl_xor(s, 16, 32);
  if (part == 0) out[(size_t)tile * 16 + m] = s + b2s[0];
}

// -------------------- launch ------------------------------------------------
extern "C" void kernel_launch(void* const* d_in, const int* in_sizes, int n_in,
                              void* d_out, int out_size, void* d_ws, size_t ws_size,
                              hipStream_t stream) {
  (void)in_sizes; (void)n_in; (void)out_size; (void)ws_size;
  const float* x      = (const float*)d_in[0];
  const float* eattr  = (const float*)d_in[1];
  const long long* ei = (const long long*)d_in[2];

  float* agg1 = (float*)((char*)d_ws + OFF_AGG1);
  float* agg2 = (float*)((char*)d_ws + OFF_AGG2);
  float* h1   = (float*)((char*)d_ws + OFF_H1);
  _Float16* wf = (_Float16*)((char*)d_ws + OFF_WF);

  // init both agg buffers (contiguous) to -inf for scatter-max
  k_fill<<<8192, 256, 0, stream>>>(agg1, (size_t)N_NODES * 128 * 2,
                                   -__builtin_huge_valf());

  // weight fragment conversion (f32 row-major -> f16 WMMA-B frags)
  k_wfrag<<<3 * 4, 512, 0, stream>>>((const float*)d_in[3],  wf + WF_P1_0, 64, 4);
  k_wfrag<<<2 * 4, 512, 0, stream>>>((const float*)d_in[5],  wf + WF_P1_1, 64, 4);
  k_wfrag<<<2 * 8, 512, 0, stream>>>((const float*)d_in[7],  wf + WF_P1_2, 128, 8);
  k_wfrag<<<5 * 4, 512, 0, stream>>>((const float*)d_in[9],  wf + WF_G1_0, 64, 4);
  k_wfrag<<<2 * 4, 512, 0, stream>>>((const float*)d_in[11], wf + WF_G1_1, 64, 4);
  k_wfrag<<<2 * 4, 512, 0, stream>>>((const float*)d_in[13], wf + WF_G1_2, 64, 4);
  k_wfrag<<<5 * 4, 512, 0, stream>>>((const float*)d_in[15], wf + WF_P2_0, 64, 4);
  k_wfrag<<<2 * 4, 512, 0, stream>>>((const float*)d_in[17], wf + WF_P2_1, 64, 4);
  k_wfrag<<<2 * 8, 512, 0, stream>>>((const float*)d_in[19], wf + WF_P2_2, 128, 8);
  k_wfrag<<<6 * 4, 512, 0, stream>>>((const float*)d_in[21], wf + WF_G2_0, 64, 4);
  k_wfrag<<<2 * 4, 512, 0, stream>>>((const float*)d_in[23], wf + WF_G2_1, 64, 4);
  k_wfrag<<<2 * 4, 512, 0, stream>>>((const float*)d_in[25], wf + WF_G2_2, 64, 4);
  k_wfrag<<<2 * 4, 512, 0, stream>>>((const float*)d_in[27], wf + WF_H_0, 64, 4);
  k_wfrag<<<2 * 4, 512, 0, stream>>>((const float*)d_in[29], wf + WF_H_1, 64, 4);

  // layer 1: edge messages -> scatter max -> node update
  k_phi<32, 96, 3><<<N_EDGES / 16 / 4, 128, 0, stream>>>(
      x, eattr, ei, wf + WF_P1_0, wf + WF_P1_1, wf + WF_P1_2,
      (const float*)d_in[4], (const float*)d_in[6], (const float*)d_in[8], agg1);
  k_gamma1<<<N_NODES / 16 / 4, 128, 0, stream>>>(
      agg1, x, wf + WF_G1_0, wf + WF_G1_1, wf + WF_G1_2,
      (const float*)d_in[10], (const float*)d_in[12], (const float*)d_in[14], h1);

  // layer 2: edge messages on h1 -> scatter max -> node update + head
  k_phi<64, 160, 5><<<N_EDGES / 16 / 4, 128, 0, stream>>>(
      h1, eattr, ei, wf + WF_P2_0, wf + WF_P2_1, wf + WF_P2_2,
      (const float*)d_in[16], (const float*)d_in[18], (const float*)d_in[20], agg2);
  k_gamma2_head<<<N_NODES / 16 / 4, 128, 0, stream>>>(
      agg2, h1, wf + WF_G2_0, wf + WF_G2_1, wf + WF_G2_2,
      (const float*)d_in[22], (const float*)d_in[24], (const float*)d_in[26],
      wf + WF_H_0, wf + WF_H_1,
      (const float*)d_in[28], (const float*)d_in[30],
      (const float*)d_in[31], (const float*)d_in[32],
      (float*)d_out);
}
